// TextCondBlock_66683662238506
// MI455X (gfx1250) — compile-verified, matrix operand order checked by reading, not run on hardware
//
#include <hip/hip_runtime.h>

typedef __bf16 bf16_t;
typedef unsigned int u32;
typedef __attribute__((ext_vector_type(16))) bf16_t v16bf;
typedef __attribute__((ext_vector_type(8)))  float  v8f;
typedef __attribute__((ext_vector_type(4)))  u32    v4u;
typedef __attribute__((ext_vector_type(4)))  int    v4i;

constexpr int kB = 8, kC = 1024, kS = 2048, kT = 512, kH = 16, kD = 64;
constexpr float kEPS = 1e-5f;

#ifndef __has_builtin
#define __has_builtin(x) 0
#endif
#if __has_builtin(__builtin_amdgcn_global_load_async_to_lds_b128) && \
    __has_builtin(__builtin_amdgcn_s_wait_asynccnt)
#define USE_ASYNC_LDS 1
#else
#define USE_ASYNC_LDS 0
#endif

#define AS1 __attribute__((address_space(1)))
#define AS3 __attribute__((address_space(3)))

#if USE_ASYNC_LDS
__device__ __forceinline__ void async_copy16(bf16_t* lds, const bf16_t* g) {
  __builtin_amdgcn_global_load_async_to_lds_b128((AS1 v4i*)g, (AS3 v4i*)lds, 0, 0);
}
#define ASYNC_WAIT(n) __builtin_amdgcn_s_wait_asynccnt(n)
#else
#define ASYNC_WAIT(n)
#endif

// ---------------------------------------------------------------- helpers
__device__ __forceinline__ bf16_t tobf(float f) {
  union { float f; u32 u; } a; a.f = f;
  u32 r = a.u + 0x7FFFu + ((a.u >> 16) & 1u);          // round-nearest-even
  union { unsigned short s; bf16_t h; } b;
  b.s = (unsigned short)(r >> 16);
  return b.h;
}

// 16-bit matrix fragment (ISA 7.12.2): lane-striped rows, K contiguous.
// Elements [0..7] = K 8g..8g+7, [8..15] = K 16+8g..23+8g  ->  two b128 loads.
__device__ __forceinline__ v16bf frag16(const bf16_t* base, int pitch, int lane) {
  const int m = lane & 15, g = lane >> 4;
  const bf16_t* r = base + m * pitch + g * 8;
  union { struct { v4u lo, hi; } p; v16bf v; } u;
  u.p.lo = *(const v4u*)(r);
  u.p.hi = *(const v4u*)(r + 16);
  return u.v;
}

__device__ __forceinline__ v8f wmma_bf16(v16bf a, v16bf b, v8f c) {
  return __builtin_amdgcn_wmma_f32_16x16x32_bf16(false, a, false, b, (short)0, c,
                                                 false, false);
}

// --------------------------------------------- W (KxN) -> Wt (NxK) bf16
__global__ __launch_bounds__(256) void wt_kernel(const float* __restrict__ W,
                                                 bf16_t* __restrict__ Wt) {
  __shared__ float t[32][33];
  const int tx = threadIdx.x & 31, ty = threadIdx.x >> 5;
  const int k0 = blockIdx.x * 32, n0 = blockIdx.y * 32;
#pragma unroll
  for (int ii = 0; ii < 4; ++ii)
    t[ty + 8 * ii][tx] = W[(size_t)(k0 + ty + 8 * ii) * kC + n0 + tx];
  __syncthreads();
#pragma unroll
  for (int ii = 0; ii < 4; ++ii)
    Wt[(size_t)(n0 + ty + 8 * ii) * kC + k0 + tx] = tobf(t[tx][ty + 8 * ii]);
}

// --------------------------------------------- LN over C of x (B,C,S) -> xn (B,S,C) bf16
__global__ __launch_bounds__(256) void ln_x_kernel(const float* __restrict__ x,
                                                   const float* __restrict__ gg,
                                                   const float* __restrict__ bb,
                                                   bf16_t* __restrict__ xn) {
  __shared__ float red[8][32];
  __shared__ float mu_s[32], rs_s[32];
  __shared__ bf16_t tile[32][34];
  const int tid = threadIdx.x;
  const int sx = tid & 31, cy = tid >> 5;
  const int b = blockIdx.y, s0 = blockIdx.x * 32;
  const float* xb = x + (size_t)b * kC * kS + (s0 + sx);

  float sum = 0.f, sq = 0.f;
  for (int c = cy; c < kC; c += 8) {
    float v = xb[(size_t)c * kS];
    sum += v; sq += v * v;
  }
  red[cy][sx] = sum; __syncthreads();
  if (cy == 0) {
    float t = 0.f;
#pragma unroll
    for (int i = 0; i < 8; ++i) t += red[i][sx];
    mu_s[sx] = t * (1.f / kC);
  }
  __syncthreads();
  red[cy][sx] = sq; __syncthreads();
  if (cy == 0) {
    float t = 0.f;
#pragma unroll
    for (int i = 0; i < 8; ++i) t += red[i][sx];
    float mu = mu_s[sx];
    rs_s[sx] = rsqrtf(t * (1.f / kC) - mu * mu + kEPS);
  }
  __syncthreads();

  for (int c0 = 0; c0 < kC; c0 += 32) {
#pragma unroll
    for (int ii = 0; ii < 4; ++ii) {
      int cc = cy + 8 * ii, c = c0 + cc;
      float v = x[(size_t)b * kC * kS + (size_t)c * kS + s0 + sx];
      tile[cc][sx] = tobf((v - mu_s[sx]) * rs_s[sx] * gg[c] + bb[c]);
    }
    __syncthreads();
    const int cn = tid & 31, sr = tid >> 5;
#pragma unroll
    for (int ii = 0; ii < 4; ++ii) {
      int s = sr + 8 * ii;
      xn[((size_t)b * kS + s0 + s) * kC + c0 + cn] = tile[cn][s];
    }
    __syncthreads();
  }
}

// --------------------------------------------- LN on row-major rows (text)
__global__ __launch_bounds__(256) void ln_rows_kernel(const float* __restrict__ src,
                                                      const float* __restrict__ gg,
                                                      const float* __restrict__ bb,
                                                      bf16_t* __restrict__ dst,
                                                      int nrows) {
  const int lane = threadIdx.x & 31, wid = threadIdx.x >> 5;
  const int row = blockIdx.x * 8 + wid;
  if (row >= nrows) return;
  const float* p = src + (size_t)row * kC;
  float vals[32];
  float sum = 0.f;
#pragma unroll
  for (int j = 0; j < 32; ++j) { vals[j] = p[j * 32 + lane]; sum += vals[j]; }
#pragma unroll
  for (int off = 16; off >= 1; off >>= 1) sum += __shfl_xor(sum, off);
  const float mu = sum * (1.f / kC);
  float sq = 0.f;
#pragma unroll
  for (int j = 0; j < 32; ++j) { float d = vals[j] - mu; sq += d * d; }
#pragma unroll
  for (int off = 16; off >= 1; off >>= 1) sq += __shfl_xor(sq, off);
  const float rs = rsqrtf(sq * (1.f / kC) + kEPS);
  bf16_t* q = dst + (size_t)row * kC;
#pragma unroll
  for (int j = 0; j < 32; ++j) {
    int c = j * 32 + lane;
    q[c] = tobf((vals[j] - mu) * rs * gg[c] + bb[c]);
  }
}

// --------------------------------------------- bf16 GEMM: out = A(MxK) @ Wt(NxK)^T + bias
// Double-buffered async LDS staging; K-step 64 (8 WMMA per wave per stage).
// MODE 0: fp32 row-major MxN            (pre-RoPE Q / K)
// MODE 1: bf16 scattered V^T (B,H,D,T)  (V)
// MODE 2: fp32 (B,C,S) = resid + acc    (output projection + residual transpose)
constexpr int MT = 64, NT = 128, KT = 64;
constexpr int GP = 72;   // GEMM LDS pitch: 144B rows, 16B aligned, conflict-free
constexpr int AP = 40;   // attention LDS pitch: 80B rows

template <int MODE>
__global__ __launch_bounds__(256) void gemm_bf16_kernel(
    const bf16_t* __restrict__ A, const bf16_t* __restrict__ Wt,
    const float* __restrict__ bias, float* __restrict__ outF,
    bf16_t* __restrict__ outB, const float* __restrict__ resid, int rowsPerB) {
  __shared__ bf16_t As[2][MT][GP];
  __shared__ bf16_t Bts[2][NT][GP];
  const int tid = threadIdx.x, lane = tid & 31, wid = tid >> 5;
  const int m0 = blockIdx.x * MT, n0 = blockIdx.y * NT;
  const int wm = (wid & 1) * 32, wn = (wid >> 1) * 32;

  v8f acc[2][2];
  {
    const float bv0 = bias[n0 + wn + (lane & 15)];
    const float bv1 = bias[n0 + wn + 16 + (lane & 15)];
#pragma unroll
    for (int i = 0; i < 8; ++i) {
      acc[0][0][i] = bv0; acc[1][0][i] = bv0;
      acc[0][1][i] = bv1; acc[1][1][i] = bv1;
    }
  }

  auto stage = [&](int s, int k0) {
#if USE_ASYNC_LDS
    // A tile: 64x64 bf16 = 8KB = 512 x 16B -> 2 per thread
#pragma unroll
    for (int i = 0; i < 2; ++i) {
      const int idx = tid + i * 256, r = idx >> 3, c = (idx & 7) * 8;
      async_copy16(&As[s][r][c], A + (size_t)(m0 + r) * kC + k0 + c);
    }
    // Bt tile: 128x64 bf16 = 16KB = 1024 x 16B -> 4 per thread
#pragma unroll
    for (int i = 0; i < 4; ++i) {
      const int idx = tid + i * 256, r = idx >> 3, c = (idx & 7) * 8;
      async_copy16(&Bts[s][r][c], Wt + (size_t)(n0 + r) * kC + k0 + c);
    }
#else
    const u32* Ag = (const u32*)A;
    const u32* Wg = (const u32*)Wt;
#pragma unroll
    for (int i = 0; i < 8; ++i) {        // A tile: 2048 u32
      int idx = tid + i * 256, r = idx >> 5, cu = idx & 31;
      *(u32*)&As[s][r][cu * 2] = Ag[(size_t)(m0 + r) * (kC / 2) + (k0 >> 1) + cu];
    }
#pragma unroll
    for (int i = 0; i < 16; ++i) {       // Bt tile: 4096 u32
      int idx = tid + i * 256, r = idx >> 5, cu = idx & 31;
      *(u32*)&Bts[s][r][cu * 2] = Wg[(size_t)(n0 + r) * (kC / 2) + (k0 >> 1) + cu];
    }
#endif
  };

  constexpr int NK = kC / KT;
  stage(0, 0);
  for (int it = 0; it < NK; ++it) {
    const int cur = it & 1;
    if (it + 1 < NK) {
      stage(1 - cur, (it + 1) * KT);
      ASYNC_WAIT(6);          // only current tile's (oldest) 6 copies must land
    } else {
      ASYNC_WAIT(0);
    }
    __syncthreads();
#pragma unroll
    for (int kk = 0; kk < KT; kk += 32) {
      v16bf af0 = frag16(&As[cur][wm][kk], GP, lane);
      v16bf af1 = frag16(&As[cur][wm + 16][kk], GP, lane);
      v16bf bf0 = frag16(&Bts[cur][wn][kk], GP, lane);
      v16bf bf1 = frag16(&Bts[cur][wn + 16][kk], GP, lane);
      acc[0][0] = wmma_bf16(af0, bf0, acc[0][0]);
      acc[0][1] = wmma_bf16(af0, bf1, acc[0][1]);
      acc[1][0] = wmma_bf16(af1, bf0, acc[1][0]);
      acc[1][1] = wmma_bf16(af1, bf1, acc[1][1]);
    }
    __syncthreads();
  }

#pragma unroll
  for (int sm = 0; sm < 2; ++sm)
#pragma unroll
    for (int sn = 0; sn < 2; ++sn)
#pragma unroll
      for (int i = 0; i < 8; ++i) {
        const int m = m0 + wm + sm * 16 + i + ((lane >> 4) << 3);
        const int n = n0 + wn + sn * 16 + (lane & 15);
        const float v = acc[sm][sn][i];
        if (MODE == 0) {
          outF[(size_t)m * kC + n] = v;
        } else if (MODE == 1) {   // V^T: (B,H,D,T)
          const int b = m / rowsPerB, t = m % rowsPerB;
          const int h = n >> 6, d = n & 63;
          outB[(((size_t)b * kH + h) * kD + d) * rowsPerB + t] = tobf(v);
        } else {
          const int b = m / rowsPerB, s = m % rowsPerB;
          const size_t idx = ((size_t)b * kC + n) * kS + s;
          outF[idx] = resid[idx] + v;
        }
      }
}

// --------------------------------------------- RoPE + bf16 convert -> (B,H,rows,D)
__global__ __launch_bounds__(256) void rope_kernel(const float* __restrict__ src,
                                                   bf16_t* __restrict__ dst, int rows,
                                                   float posScale, float outScale) {
  const size_t total = (size_t)kB * rows * kH * 32;
  for (size_t idx = (size_t)blockIdx.x * blockDim.x + threadIdx.x; idx < total;
       idx += (size_t)gridDim.x * blockDim.x) {
    const int d = (int)(idx & 31);
    size_t r1 = idx >> 5;
    const int hh = (int)(r1 % kH); r1 /= kH;
    const int sr = (int)(r1 % rows);
    const int b = (int)(r1 / rows);
    const float* p = src + ((size_t)b * rows + sr) * kC + hh * kD;
    const float t1 = p[d], t2 = p[d + 32];
    const float inv = __powf(10000.f, -(float)d * (1.f / 32.f));
    float s, c;
    __sincosf((float)sr * posScale * inv, &s, &c);
    bf16_t* q = dst + (((size_t)b * kH + hh) * rows + sr) * kD;
    q[d]      = tobf((t1 * c - t2 * s) * outScale);
    q[d + 32] = tobf((t1 * s + t2 * c) * outScale);
  }
}

// --------------------------------------------- flash attention per (b,h)
constexpr int KP = 72;   // Ks pitch

__global__ __launch_bounds__(256) void attn_kernel(const bf16_t* __restrict__ q,
                                                   const bf16_t* __restrict__ k,
                                                   const bf16_t* __restrict__ vt,
                                                   bf16_t* __restrict__ o) {
  __shared__ bf16_t Ks[2][32][KP];       // K tile, rows t, cols d
  __shared__ bf16_t Vts[2][kD][AP];      // V^T tile, rows d, cols t
  __shared__ bf16_t Ps[8][16][AP];       // per-wave P tile
  const int tid = threadIdx.x, lane = tid & 31, wid = tid >> 5;
  const int bh = blockIdx.y, b = bh / kH, h = bh % kH;
  const int s0 = blockIdx.x * 128 + wid * 16;
  const int g = lane >> 4;

  // Q fragments (held for whole T loop); pairs contiguous in d -> b128 loads.
  v16bf qf[2];
  {
    const bf16_t* qrow = q + ((size_t)bh * kS + s0 + (lane & 15)) * kD;
#pragma unroll
    for (int kd = 0; kd < 2; ++kd) {
      union { struct { v4u lo, hi; } p; v16bf v; } u;
      u.p.lo = *(const v4u*)(qrow + kd * 32 + g * 8);
      u.p.hi = *(const v4u*)(qrow + kd * 32 + g * 8 + 16);
      qf[kd] = u.v;
    }
  }

  float mrow[8], lrow[8];
  v8f oacc[4];
#pragma unroll
  for (int i = 0; i < 8; ++i) { mrow[i] = -3.0e38f; lrow[i] = 0.f; }
#pragma unroll
  for (int f = 0; f < 4; ++f) oacc[f] = v8f{};

  const bf16_t* kb = k + (size_t)bh * kT * kD;
  const bf16_t* vb = vt + (size_t)bh * kD * kT;

  auto stage = [&](int s, int t0) {
#if USE_ASYNC_LDS
    const int rk = tid >> 3, ck = (tid & 7) * 8;        // 32 x 64
    async_copy16(&Ks[s][rk][ck], kb + (size_t)(t0 + rk) * kD + ck);
    const int rv = tid >> 2, cv = (tid & 3) * 8;        // 64 x 32
    async_copy16(&Vts[s][rv][cv], vb + (size_t)rv * kT + t0 + cv);
#else
    const u32* kg = (const u32*)kb;
    const u32* vg = (const u32*)vb;
#pragma unroll
    for (int i = 0; i < 4; ++i) {
      int idx = tid + i * 256;
      int rk = idx >> 5, ck = idx & 31;
      *(u32*)&Ks[s][rk][ck * 2] = kg[(size_t)(t0 + rk) * (kD / 2) + ck];
      int rv = idx >> 4, cv = idx & 15;
      *(u32*)&Vts[s][rv][cv * 2] = vg[(size_t)rv * (kT / 2) + (t0 >> 1) + cv];
    }
#endif
  };

  constexpr int NSTEP = kT / 32;
  stage(0, 0);
  for (int it = 0; it < NSTEP; ++it) {
    const int cur = it & 1;
    if (it + 1 < NSTEP) {
      stage(1 - cur, (it + 1) * 32);
      ASYNC_WAIT(2);
    } else {
      ASYNC_WAIT(0);
    }
    __syncthreads();

    // scores: 16 x 32 (two N-subtiles), contraction over d in two K-steps.
    v8f sc[2]; sc[0] = v8f{}; sc[1] = v8f{};
#pragma unroll
    for (int kd = 0; kd < 2; ++kd)
#pragma unroll
      for (int sn = 0; sn < 2; ++sn)
        sc[sn] = wmma_bf16(qf[kd], frag16(&Ks[cur][sn * 16][kd * 32], KP, lane),
                           sc[sn]);

    // online softmax
    float alpha[8];
#pragma unroll
    for (int i = 0; i < 8; ++i) {
      float vmax = fmaxf(sc[0][i], sc[1][i]);
#pragma unroll
      for (int off = 8; off >= 1; off >>= 1) vmax = fmaxf(vmax, __shfl_xor(vmax, off));
      const float mnew = fmaxf(mrow[i], vmax);
      alpha[i] = __expf(mrow[i] - mnew);
      mrow[i] = mnew;
      float p0 = __expf(sc[0][i] - mnew);
      float p1 = __expf(sc[1][i] - mnew);
      const int prow = i + (g << 3);
      Ps[wid][prow][(lane & 15)]      = tobf(p0);
      Ps[wid][prow][16 + (lane & 15)] = tobf(p1);
      float s = p0 + p1;
#pragma unroll
      for (int off = 8; off >= 1; off >>= 1) s += __shfl_xor(s, off);
      lrow[i] = lrow[i] * alpha[i] + s;
    }
#pragma unroll
    for (int f = 0; f < 4; ++f)
#pragma unroll
      for (int i = 0; i < 8; ++i) oacc[f][i] *= alpha[i];

    // PV: A = P (16x32 over t), B = V^T tile (lane-striped d, contiguous t)
    v16bf pf = frag16(&Ps[wid][0][0], AP, lane);
#pragma unroll
    for (int dn = 0; dn < 4; ++dn)
      oacc[dn] = wmma_bf16(pf, frag16(&Vts[cur][dn * 16][0], AP, lane), oacc[dn]);
    __syncthreads();
  }

  // normalize and write o (B,S,C) bf16
#pragma unroll
  for (int dn = 0; dn < 4; ++dn)
#pragma unroll
    for (int i = 0; i < 8; ++i) {
      const int s = s0 + i + (g << 3);
      const int c = h * kD + dn * 16 + (lane & 15);
      o[((size_t)b * kS + s) * kC + c] = tobf(oacc[dn][i] / lrow[i]);
    }
}

// ---------------------------------------------------------------- launch
extern "C" void kernel_launch(void* const* d_in, const int* in_sizes, int n_in,
                              void* d_out, int out_size, void* d_ws, size_t ws_size,
                              hipStream_t stream) {
  const float* x   = (const float*)d_in[0];
  const float* te  = (const float*)d_in[1];
  const float* g_x = (const float*)d_in[2];
  const float* b_x = (const float*)d_in[3];
  const float* g_t = (const float*)d_in[4];
  const float* b_t = (const float*)d_in[5];
  const float* Wq  = (const float*)d_in[6];
  const float* bq  = (const float*)d_in[7];
  const float* Wk  = (const float*)d_in[8];
  const float* bk  = (const float*)d_in[9];
  const float* Wv  = (const float*)d_in[10];
  const float* bv  = (const float*)d_in[11];
  const float* Wo  = (const float*)d_in[12];
  const float* bo  = (const float*)d_in[13];
  float* out = (float*)d_out;
  char* ws = (char*)d_ws;

  size_t off = 0;
  auto alloc = [&](size_t bytes) {
    size_t o = off; off += (bytes + 255) & ~(size_t)255; return o;
  };
  bf16_t* xn   = (bf16_t*)(ws + alloc((size_t)kB * kS * kC * 2));
  bf16_t* tn   = (bf16_t*)(ws + alloc((size_t)kB * kT * kC * 2));
  bf16_t* Wqt  = (bf16_t*)(ws + alloc((size_t)kC * kC * 2));
  bf16_t* Wkt  = (bf16_t*)(ws + alloc((size_t)kC * kC * 2));
  bf16_t* Wvt  = (bf16_t*)(ws + alloc((size_t)kC * kC * 2));
  bf16_t* Wot  = (bf16_t*)(ws + alloc((size_t)kC * kC * 2));
  float*  qraw = (float*)(ws + alloc((size_t)kB * kS * kC * 4));
  float*  kraw = (float*)(ws + alloc((size_t)kB * kT * kC * 4));
  bf16_t* qbf  = (bf16_t*)(ws + alloc((size_t)kB * kS * kC * 2));
  bf16_t* kbf  = (bf16_t*)(ws + alloc((size_t)kB * kT * kC * 2));
  bf16_t* vtbf = (bf16_t*)(ws + alloc((size_t)kB * kT * kC * 2));
  bf16_t* obf  = (bf16_t*)qraw;   // alias: qraw dead after rope, before attn

  wt_kernel<<<dim3(kC / 32, kC / 32), 256, 0, stream>>>(Wq, Wqt);
  wt_kernel<<<dim3(kC / 32, kC / 32), 256, 0, stream>>>(Wk, Wkt);
  wt_kernel<<<dim3(kC / 32, kC / 32), 256, 0, stream>>>(Wv, Wvt);
  wt_kernel<<<dim3(kC / 32, kC / 32), 256, 0, stream>>>(Wo, Wot);

  ln_x_kernel<<<dim3(kS / 32, kB), 256, 0, stream>>>(x, g_x, b_x, xn);
  ln_rows_kernel<<<(kB * kT) / 8, 256, 0, stream>>>(te, g_t, b_t, tn, kB * kT);

  gemm_bf16_kernel<0><<<dim3((kB * kS) / MT, kC / NT), 256, 0, stream>>>(
      xn, Wqt, bq, qraw, nullptr, nullptr, kS);
  gemm_bf16_kernel<0><<<dim3((kB * kT) / MT, kC / NT), 256, 0, stream>>>(
      tn, Wkt, bk, kraw, nullptr, nullptr, kT);
  gemm_bf16_kernel<1><<<dim3((kB * kT) / MT, kC / NT), 256, 0, stream>>>(
      tn, Wvt, bv, nullptr, vtbf, nullptr, kT);

  rope_kernel<<<8192, 256, 0, stream>>>(qraw, qbf, kS, 1.0f, 0.125f);   // q * 1/sqrt(D)
  rope_kernel<<<2048, 256, 0, stream>>>(kraw, kbf, kT, (float)kS / kT, 1.0f);

  attn_kernel<<<dim3(kS / 128, kB * kH), 256, 0, stream>>>(qbf, kbf, vtbf, obf);

  gemm_bf16_kernel<2><<<dim3((kB * kS) / MT, kC / NT), 256, 0, stream>>>(
      obf, Wot, bo, out, nullptr, x, kS);
}
